// PointNetFeaturePropagation_46969762349059
// MI455X (gfx1250) — matrix operand order repaired and checked by reading, not compiled, not run on hardware
//
#include <hip/hip_runtime.h>
#include <hip/hip_bf16.h>

typedef __attribute__((ext_vector_type(16))) _Float16 v16h;
typedef __attribute__((ext_vector_type(8)))  float    v8f;
typedef __attribute__((ext_vector_type(4)))  float    v4f;

#define BATCH 16
#define NN1   4096
#define NN2   1024
#define CC1   128
#define CC2   256
#define INCH  384
#define COUT  256
#define MROWS (BATCH * NN1)   // 65536

// ---------------------------------------------------------------------------
// Stage 1: 3-NN interpolation + concat[points1 | interp] -> x as f16 (M x 384)
// One block = 256 points of one batch. Phase A: per-thread top-3 over LDS xyz2.
// Phase B: thread == channel, coalesced gather of points2.
// ---------------------------------------------------------------------------
__global__ __launch_bounds__(256) void interp_concat_kernel(
    const float* __restrict__ xyz1, const float* __restrict__ xyz2,
    const float* __restrict__ points1, const float* __restrict__ points2,
    _Float16* __restrict__ xh)
{
    __shared__ float s_xyz[NN2 * 3];
    __shared__ int   s_idx[256][3];
    __shared__ float s_w[256][3];

    const int b    = blockIdx.x >> 4;     // 16 tiles per batch
    const int tile = blockIdx.x & 15;
    const int t    = threadIdx.x;

    for (int i = t; i < NN2 * 3; i += 256)
        s_xyz[i] = xyz2[(size_t)b * NN2 * 3 + i];
    __syncthreads();

    // Phase A: this thread's point
    const int n1 = tile * 256 + t;
    const size_t prow = (size_t)b * NN1 + n1;
    const float px = xyz1[prow * 3 + 0];
    const float py = xyz1[prow * 3 + 1];
    const float pz = xyz1[prow * 3 + 2];

    float d0 = 3.4e38f, d1 = 3.4e38f, d2 = 3.4e38f;
    int   i0 = 0, i1 = 0, i2 = 0;
    for (int j = 0; j < NN2; ++j) {
        const float dx = px - s_xyz[j * 3 + 0];
        const float dy = py - s_xyz[j * 3 + 1];
        const float dz = pz - s_xyz[j * 3 + 2];
        const float d  = dx * dx + dy * dy + dz * dz;
        if (d < d2) {
            if (d < d1) {
                if (d < d0) { d2 = d1; i2 = i1; d1 = d0; i1 = i0; d0 = d; i0 = j; }
                else        { d2 = d1; i2 = i1; d1 = d;  i1 = j; }
            } else          { d2 = d;  i2 = j; }
        }
    }
    float w0 = 1.0f / (d0 + 1e-8f);
    float w1 = 1.0f / (d1 + 1e-8f);
    float w2 = 1.0f / (d2 + 1e-8f);
    const float inv = 1.0f / (w0 + w1 + w2);
    s_idx[t][0] = i0; s_idx[t][1] = i1; s_idx[t][2] = i2;
    s_w[t][0] = w0 * inv; s_w[t][1] = w1 * inv; s_w[t][2] = w2 * inv;
    __syncthreads();

    // Phase B: thread t = channel; loop over the block's 256 points
    const float* __restrict__ P2 = points2 + (size_t)b * NN2 * CC2;
    for (int p = 0; p < 256; ++p) {
        const size_t row = (size_t)b * NN1 + tile * 256 + p;
        const int j0 = s_idx[p][0], j1 = s_idx[p][1], j2 = s_idx[p][2];
        const float a0 = s_w[p][0], a1 = s_w[p][1], a2 = s_w[p][2];
        const float v = a0 * P2[(size_t)j0 * CC2 + t]
                      + a1 * P2[(size_t)j1 * CC2 + t]
                      + a2 * P2[(size_t)j2 * CC2 + t];
        xh[row * INCH + CC1 + t] = (_Float16)v;
        if (t < CC1)
            xh[row * INCH + t] = (_Float16)points1[row * CC1 + t];
    }
}

// ---------------------------------------------------------------------------
// Stage 2: GEMM  Y[M,256] = X[M,K](f16) * W[256,K]^T(f16) + bias  (f32 accum)
// One wave = 16-row M-tile x 128-col N-half (8 wmma accum tiles = 64 VGPRs),
// keeping the wave under 256 VGPRs for full direct addressing + occupancy.
// A rows re-read by the second N-half wave hit L2 (x buffer is L2-resident).
// ---------------------------------------------------------------------------
__global__ __launch_bounds__(256) void gemm_wmma_kernel(
    const _Float16* __restrict__ X, const _Float16* __restrict__ W,
    const float* __restrict__ bias, float* __restrict__ Y, int K)
{
    const int lane  = threadIdx.x & 31;
    const int wid   = (blockIdx.x * blockDim.x + threadIdx.x) >> 5;
    const int nhalf = wid & 1;              // which 128-wide N half
    const int row0  = (wid >> 1) * 16;      // M tile
    const int r     = lane & 15;            // M (A) / N-in-tile (B) across lanes
    const int h     = lane >> 4;            // lane half selects K chunk

    v8f acc[8];
#pragma unroll
    for (int j = 0; j < 8; ++j) acc[j] = (v8f)(0.0f);

    // A layout (16-bit 16x32): lane holds K = h*8..h*8+7 and 16+h*8..16+h*8+7
    const _Float16* xp = X + (size_t)(row0 + r) * K + h * 8;
    // B base: N rows for this wave start at nhalf*128
    const _Float16* wb = W + (size_t)(nhalf * 128 + r) * K + h * 8;

    for (int k0 = 0; k0 < K; k0 += 32) {
        v16h a;
        ((v4f*)&a)[0] = *(const v4f*)(xp + k0);
        ((v4f*)&a)[1] = *(const v4f*)(xp + k0 + 16);
#pragma unroll
        for (int j = 0; j < 8; ++j) {
            // B layout (16-bit 32x16): lane (N = r) holds K = h*8..h*8+7 and 16+h*8..16+h*8+7
            const _Float16* wp = wb + (size_t)(j * 16) * K + k0;
            v16h bt;
            ((v4f*)&bt)[0] = *(const v4f*)(wp);
            ((v4f*)&bt)[1] = *(const v4f*)(wp + 16);
            acc[j] = __builtin_amdgcn_wmma_f32_16x16x32_f16(
                false, a, false, bt, (short)0, acc[j], false, false);
        }
    }

    // C/D layout: VGPR rr -> M = rr + 8*h, N = r
    const int mh = h * 8;
    const int ncol0 = nhalf * 128;
#pragma unroll
    for (int j = 0; j < 8; ++j) {
        const float bv = bias[ncol0 + j * 16 + r];
#pragma unroll
        for (int rr = 0; rr < 8; ++rr) {
            Y[(size_t)(row0 + mh + rr) * COUT + ncol0 + j * 16 + r] = acc[j][rr] + bv;
        }
    }
}

// ---------------------------------------------------------------------------
// Stage 3: per-channel batch stats (deterministic 2-stage reduction)
// ---------------------------------------------------------------------------
__global__ __launch_bounds__(256) void col_stats_partial_kernel(
    const float* __restrict__ y, float* __restrict__ partial)
{
    const int p = blockIdx.x;   // 256 row-partitions of 256 rows each
    const int t = threadIdx.x;  // channel
    const float* yp = y + (size_t)p * 256 * COUT;
    float s = 0.0f, ss = 0.0f;
    for (int rr = 0; rr < 256; ++rr) {
        const float v = yp[(size_t)rr * COUT + t];
        s += v; ss += v * v;
    }
    partial[p * 512 + t]       = s;
    partial[p * 512 + 256 + t] = ss;
}

__global__ __launch_bounds__(256) void col_stats_final_kernel(
    const float* __restrict__ partial, const float* __restrict__ g,
    const float* __restrict__ be, float* __restrict__ stats)
{
    const int c = threadIdx.x;
    float s = 0.0f, ss = 0.0f;
    for (int p = 0; p < 256; ++p) {
        s  += partial[p * 512 + c];
        ss += partial[p * 512 + 256 + c];
    }
    const float mean = s * (1.0f / (float)MROWS);
    const float var  = ss * (1.0f / (float)MROWS) - mean * mean;
    const float sc   = rsqrtf(var + 1e-5f) * g[c];
    stats[c]        = sc;                 // scale
    stats[256 + c]  = be[c] - mean * sc;  // shift
}

// ---------------------------------------------------------------------------
// Stage 4: BN + ReLU epilogues
// ---------------------------------------------------------------------------
__global__ __launch_bounds__(256) void bn_relu_f16_kernel(
    const float* __restrict__ y, const float* __restrict__ stats,
    _Float16* __restrict__ xo)
{
    const size_t i = (size_t)blockIdx.x * 256 + threadIdx.x;
    const int c = threadIdx.x;  // row-major, 256 ch => lane == channel
    const float v = fmaf(y[i], stats[c], stats[256 + c]);
    xo[i] = (_Float16)fmaxf(v, 0.0f);
}

__global__ __launch_bounds__(256) void bn_relu_f32_kernel(
    const float* __restrict__ y, const float* __restrict__ stats,
    float* __restrict__ out)
{
    const size_t i = (size_t)blockIdx.x * 256 + threadIdx.x;
    const int c = threadIdx.x;
    const float v = fmaf(y[i], stats[c], stats[256 + c]);
    out[i] = fmaxf(v, 0.0f);
}

// ---------------------------------------------------------------------------
// Weight conversion fp32 -> f16
// ---------------------------------------------------------------------------
__global__ __launch_bounds__(256) void cvt_f16_kernel(
    const float* __restrict__ w, _Float16* __restrict__ wh, int n)
{
    const int i = blockIdx.x * 256 + threadIdx.x;
    if (i < n) wh[i] = (_Float16)w[i];
}

// ---------------------------------------------------------------------------
extern "C" void kernel_launch(void* const* d_in, const int* in_sizes, int n_in,
                              void* d_out, int out_size, void* d_ws, size_t ws_size,
                              hipStream_t stream) {
    const float* xyz1    = (const float*)d_in[0];
    const float* xyz2    = (const float*)d_in[1];
    const float* points1 = (const float*)d_in[2];
    const float* points2 = (const float*)d_in[3];
    const float* w0 = (const float*)d_in[4];
    const float* b0 = (const float*)d_in[5];
    const float* g0 = (const float*)d_in[6];
    const float* be0 = (const float*)d_in[7];
    const float* w1 = (const float*)d_in[8];
    const float* b1 = (const float*)d_in[9];
    const float* g1 = (const float*)d_in[10];
    const float* be1 = (const float*)d_in[11];
    float* out = (float*)d_out;

    // workspace layout
    char* ws = (char*)d_ws;
    const size_t XH_BYTES = (size_t)MROWS * INCH * sizeof(_Float16);  // 48 MB (also reused for 256-wide x1)
    const size_t Y_BYTES  = (size_t)MROWS * COUT * sizeof(float);     // 64 MB
    _Float16* xh   = (_Float16*)ws;
    float*    ybuf = (float*)(ws + XH_BYTES);
    _Float16* w0h  = (_Float16*)(ws + XH_BYTES + Y_BYTES);
    _Float16* w1h  = (_Float16*)(ws + XH_BYTES + Y_BYTES + (size_t)COUT * INCH * 2);
    float* partial = (float*)(ws + XH_BYTES + Y_BYTES + (size_t)COUT * INCH * 2
                                   + (size_t)COUT * COUT * 2);
    float* stats   = (float*)((char*)partial + 256 * 512 * sizeof(float));

    // weights -> f16
    cvt_f16_kernel<<<(COUT * INCH + 255) / 256, 256, 0, stream>>>(w0, w0h, COUT * INCH);
    cvt_f16_kernel<<<(COUT * COUT + 255) / 256, 256, 0, stream>>>(w1, w1h, COUT * COUT);

    // 3-NN interpolate + concat -> x (f16, M x 384)
    interp_concat_kernel<<<BATCH * 16, 256, 0, stream>>>(xyz1, xyz2, points1, points2, xh);

    // layer 0: GEMM (K=384) -> stats -> BN+ReLU -> x1 (f16, M x 256, reuses xh buffer)
    // waves = (M/16) * 2 N-halves = 8192; 8 waves per block -> 1024 blocks
    gemm_wmma_kernel<<<(MROWS / 16) * 2 / 8, 256, 0, stream>>>(xh, w0h, b0, ybuf, INCH);
    col_stats_partial_kernel<<<256, 256, 0, stream>>>(ybuf, partial);
    col_stats_final_kernel<<<1, 256, 0, stream>>>(partial, g0, be0, stats);
    bn_relu_f16_kernel<<<MROWS, 256, 0, stream>>>(ybuf, stats, xh);

    // layer 1: GEMM (K=256) -> stats -> BN+ReLU -> out (fp32)
    gemm_wmma_kernel<<<(MROWS / 16) * 2 / 8, 256, 0, stream>>>(xh, w1h, b1, ybuf, COUT);
    col_stats_partial_kernel<<<256, 256, 0, stream>>>(ybuf, partial);
    col_stats_final_kernel<<<1, 256, 0, stream>>>(partial, g1, be1, stats);
    bn_relu_f32_kernel<<<MROWS, 256, 0, stream>>>(ybuf, stats, out);

    (void)in_sizes; (void)n_in; (void)out_size; (void)ws_size;
}
